// Qwen3OmniAttention_44598940401867
// MI455X (gfx1250) — matrix operand order repaired and checked by reading, not compiled, not run on hardware
//
#include <hip/hip_runtime.h>

// ---- problem constants (from reference) ----
constexpr int S = 2048, Mdim = 2048, U = 32, V = 8, H = 128;
constexpr float EPS = 1e-6f;
constexpr float SCALE = 0.08838834764831845f;   // 1/sqrt(128)
constexpr float L2BASE = 19.93156856932417f;    // log2(1e6)

typedef __attribute__((ext_vector_type(16))) __bf16 v16bf;
typedef __attribute__((ext_vector_type(8)))  __bf16 v8bf;
typedef __attribute__((ext_vector_type(8)))  float  v8f;
typedef int v4i_ll __attribute__((vector_size(16)));   // matches builtin param type

union BF16x16 { v16bf v; v8bf h[2]; };

static __device__ __forceinline__ v8f wmma_bf16(v16bf a, v16bf b, v8f c) {
  return __builtin_amdgcn_wmma_f32_16x16x32_bf16(false, a, false, b, (short)0, c,
                                                 false, false);
}

// ---- gfx1250 async global->LDS copy (ASYNCcnt-tracked), 16 bytes per lane ----
static __device__ __forceinline__ void async_ld16(const void* g, void* l) {
#if __has_builtin(__builtin_amdgcn_global_load_async_to_lds_b128)
  __builtin_amdgcn_global_load_async_to_lds_b128(
      (__attribute__((address_space(1))) v4i_ll*)(g),
      (__attribute__((address_space(3))) v4i_ll*)(l), 0, 0);
#else
  unsigned loff = (unsigned)(size_t)(__attribute__((address_space(3))) char*)(l);
  unsigned long long ga = (unsigned long long)(size_t)g;
  asm volatile("global_load_async_to_lds_b128 %0, %1, off"
               :: "v"(loff), "v"(ga) : "memory");
#endif
}

template <int N>
static __device__ __forceinline__ void wait_async() {
#if __has_builtin(__builtin_amdgcn_s_wait_asynccnt)
  __builtin_amdgcn_s_wait_asynccnt((unsigned short)N);
#else
  if (N == 0) asm volatile("s_wait_asynccnt 0x0" ::: "memory");
  else        asm volatile("s_wait_asynccnt 0x4" ::: "memory");
#endif
}

// ---------------- fp32 -> bf16 convert ----------------
__global__ __launch_bounds__(256) void cvt_bf16_kernel(const float* __restrict__ in,
                                                       __bf16* __restrict__ out) {
  int i = blockIdx.x * 256 + threadIdx.x;
  out[i] = (__bf16)in[i];
}

// ---------------- weight repack into WMMA B-fragment layout ----------------
// W is (K x N) row-major fp32.  Packed: tile (kt,nt) of 32x16 -> 32 lanes x 16
// contiguous bf16; lane g*16+n element e holds B[kt*32+g*16+e][nt*16+n].
__global__ __launch_bounds__(256) void pack_w_kernel(const float* __restrict__ W,
                                                     __bf16* __restrict__ P,
                                                     int K, int N) {
  int t = blockIdx.x * 256 + threadIdx.x;
  int lane = t & 31;
  int tile = t >> 5;
  int KT = K >> 5;
  int kt = tile % KT;
  int nt = tile / KT;
  int g = lane >> 4, n = lane & 15;
  int col = nt * 16 + n;
  size_t ob = ((size_t)tile * 32 + lane) * 16;
#pragma unroll
  for (int e = 0; e < 16; ++e) {
    int k = kt * 32 + g * 16 + e;
    P[ob + e] = (__bf16)W[(size_t)k * N + col];
  }
}

// ---------------- bf16 WMMA GEMM with async-LDS A staging ----------------
// C(MxN fp32) = A(MxK bf16 rowmajor) * Bpacked.  128 threads = 4 waves (2x2);
// each wave computes 64x64 (16 accumulators).  Block A tile (128x32) is staged
// into LDS with global_load_async_to_lds_b128, double-buffered over K.
__global__ __launch_bounds__(128) void gemm_bf16_kernel(const __bf16* __restrict__ A,
                                                        const __bf16* __restrict__ Bp,
                                                        float* __restrict__ C,
                                                        int K, int N) {
  __shared__ __align__(16) __bf16 lds_a[2][128 * 32];
  const int tid = threadIdx.x;
  const int lane = tid & 31;
  const int wave = tid >> 5;
  const int wm = wave >> 1, wn = wave & 1;
  const int mblk = blockIdx.x * 128;
  const int n0 = blockIdx.y * 128 + wn * 64;
  const int half = lane >> 4, ln = lane & 15;
  const int KT = K >> 5;

  // issue async copy of A tile kt into LDS buffer buf (8KB = 4 x 16B per thread)
  auto issue_tile = [&](int kt, int buf) {
#pragma unroll
    for (int i = 0; i < 4; ++i) {
      int c = tid + 128 * i;            // chunk id: row = c>>2, part = c&3
      int row = c >> 2, part = c & 3;
      const __bf16* g = A + (size_t)(mblk + row) * K + kt * 32 + part * 8;
      async_ld16(g, &lds_a[buf][(size_t)c * 8]);
    }
  };

  v8f acc[4][4];
#pragma unroll
  for (int i = 0; i < 4; ++i)
#pragma unroll
    for (int j = 0; j < 4; ++j)
#pragma unroll
      for (int r = 0; r < 8; ++r) acc[i][j][r] = 0.0f;

  issue_tile(0, 0);
  for (int kt = 0; kt < KT; ++kt) {
    const int cur = kt & 1;
    if (kt + 1 < KT) {
      issue_tile(kt + 1, cur ^ 1);
      wait_async<4>();   // in-order: tile kt complete, next tile may be in flight
    } else {
      wait_async<0>();
    }
    __syncthreads();

    v16bf af[4], bfr[4];
#pragma unroll
    for (int mi = 0; mi < 4; ++mi) {
      // A fragment from LDS: lanes 0-15 -> K 0-15, 16-31 -> K 16-31
      const __bf16* p = &lds_a[cur][(wm * 64 + mi * 16 + ln) * 32 + half * 16];
      BF16x16 t;
      t.h[0] = *(const v8bf*)p;
      t.h[1] = *(const v8bf*)(p + 8);
      af[mi] = t.v;
    }
#pragma unroll
    for (int ni = 0; ni < 4; ++ni) {
      const __bf16* p = Bp + ((size_t)((n0 / 16 + ni) * KT + kt) * 32 + lane) * 16;
      bfr[ni] = *(const v16bf*)p;   // 32B contiguous, aligned
    }
    if (kt + 4 < KT) {  // uniform; gfx1250 global_prefetch_b8 on the B stream
      __builtin_prefetch(Bp + ((size_t)((n0 / 16) * KT + kt + 4) * 32 + lane) * 16, 0, 1);
    }
#pragma unroll
    for (int mi = 0; mi < 4; ++mi)
#pragma unroll
      for (int ni = 0; ni < 4; ++ni)
        acc[mi][ni] = wmma_bf16(af[mi], bfr[ni], acc[mi][ni]);
    __syncthreads();   // protect LDS buf before it is overwritten next iteration
  }
#pragma unroll
  for (int mi = 0; mi < 4; ++mi)
#pragma unroll
    for (int ni = 0; ni < 4; ++ni)
#pragma unroll
      for (int r = 0; r < 8; ++r)
        C[(size_t)(mblk + wm * 64 + mi * 16 + r + 8 * half) * N + n0 + ni * 16 + ln] =
            acc[mi][ni][r];
}

// ---------------- fused RMS-norm + mRoPE ----------------
__global__ __launch_bounds__(256) void qk_prep_kernel(const float* __restrict__ in,
                                                      __bf16* __restrict__ out,
                                                      const float* __restrict__ nw,
                                                      const int* __restrict__ pos,
                                                      int nheads, int is_k) {
  const int lane = threadIdx.x & 31;
  const int wave = threadIdx.x >> 5;
  const int row = blockIdx.x * 8 + wave;
  const int s = row / nheads;
  const int hidx = row % nheads;
  const float* x = in + (size_t)row * H;
  const int h2 = lane * 2;
  float x0 = x[h2], x1 = x[h2 + 1], x2 = x[h2 + 64], x3 = x[h2 + 65];
  float ss = x0 * x0 + x1 * x1 + x2 * x2 + x3 * x3;
#pragma unroll
  for (int o = 1; o < 32; o <<= 1) ss += __shfl_xor(ss, o, 32);
  float rs = rsqrtf(ss * (1.0f / H) + EPS);
  float n0 = x0 * rs * nw[h2], n1 = x1 * rs * nw[h2 + 1];
  float n2 = x2 * rs * nw[h2 + 64], n3 = x3 * rs * nw[h2 + 65];
  int st0 = (h2 < 24) ? 0 : ((h2 < 44) ? 1 : 2);
  int st1 = (h2 + 1 < 24) ? 0 : ((h2 + 1 < 44) ? 1 : 2);
  float f0 = exp2f(-((float)h2 * (1.0f / 64.0f)) * L2BASE);
  float f1 = exp2f(-((float)(h2 + 1) * (1.0f / 64.0f)) * L2BASE);
  float a0 = (float)pos[st0 * S + s] * f0;
  float a1 = (float)pos[st1 * S + s] * f1;
  float c0, s0, c1, s1;
  __sincosf(a0, &s0, &c0);
  __sincosf(a1, &s1, &c1);
  float y0 = n0 * c0 - n2 * s0;
  float y2 = n2 * c0 + n0 * s0;
  float y1 = n1 * c1 - n3 * s1;
  float y3 = n3 * c1 + n1 * s1;
  size_t ob = is_k ? ((size_t)hidx * S + s) * (size_t)H : (size_t)row * H;
  out[ob + h2] = (__bf16)y0;
  out[ob + h2 + 1] = (__bf16)y1;
  out[ob + h2 + 64] = (__bf16)y2;
  out[ob + h2 + 65] = (__bf16)y3;
}

// ---------------- V transpose: (S, V*H) fp32 -> (V, H, S) bf16 ----------------
__global__ __launch_bounds__(256) void v_prep_kernel(const float* __restrict__ in,
                                                     __bf16* __restrict__ out) {
  int i = blockIdx.x * 256 + threadIdx.x;
  int s = i / (V * H);
  int r = i % (V * H);
  int vv = r / H, h = r % H;
  out[((size_t)vv * H + h) * S + s] = (__bf16)in[i];
}

// ---------------- flash attention: online softmax, 64-key blocks -------------
// block = 128 threads (4 waves); each wave owns 16 query rows of one head.
__global__ __launch_bounds__(128) void flash_attn_kernel(
    const __bf16* __restrict__ qb,   // (S, U, H) bf16
    const __bf16* __restrict__ kbp,  // (V, S, H) bf16
    const __bf16* __restrict__ vtb,  // (V, H, S) bf16
    __bf16* __restrict__ attnb) {    // (S, U*H) bf16
  __shared__ __align__(32) __bf16 lds_p[4][16 * 64];
  const int lane = threadIdx.x & 31;
  const int wave = threadIdx.x >> 5;
  const int u = blockIdx.y;
  const int vh = u >> 2;  // GROUPS = 4
  const int q0 = blockIdx.x * 64 + wave * 16;
  const int half = lane >> 4, ln = lane & 15;

  // Q fragments: 4 K-chunks over H=128
  v16bf qf[4];
#pragma unroll
  for (int hc = 0; hc < 4; ++hc) {
    const __bf16* p = qb + ((size_t)(q0 + ln) * U + u) * H + hc * 32 + half * 16;
    BF16x16 t;
    t.h[0] = *(const v8bf*)p;
    t.h[1] = *(const v8bf*)(p + 8);
    qf[hc] = t.v;
  }

  v8f accO[8];
#pragma unroll
  for (int i = 0; i < 8; ++i)
#pragma unroll
    for (int r = 0; r < 8; ++r) accO[i][r] = 0.0f;
  float m_i[8], l_i[8];
#pragma unroll
  for (int r = 0; r < 8; ++r) { m_i[r] = -3.0e38f; l_i[r] = 0.0f; }

  for (int kbase = 0; kbase < S; kbase += 64) {
    // ---- scores: four 16-key subtiles, K=H via 4 WMMAs each (16 WMMAs) ----
    v8f sc[4];
#pragma unroll
    for (int t = 0; t < 4; ++t)
#pragma unroll
      for (int r = 0; r < 8; ++r) sc[t][r] = 0.0f;
#pragma unroll
    for (int hc = 0; hc < 4; ++hc) {
#pragma unroll
      for (int t = 0; t < 4; ++t) {
        const __bf16* kp =
            kbp + ((size_t)vh * S + kbase + t * 16 + ln) * H + hc * 32 + half * 16;
        v16bf b = *(const v16bf*)kp;
        sc[t] = wmma_bf16(qf[hc], b, sc[t]);
      }
    }
    // ---- online softmax (state per 16-lane half; row = r + 8*half) ----
    float p[4][8];
#pragma unroll
    for (int r = 0; r < 8; ++r) {
      float x0 = sc[0][r] * SCALE, x1 = sc[1][r] * SCALE;
      float x2 = sc[2][r] * SCALE, x3 = sc[3][r] * SCALE;
      float bm = fmaxf(fmaxf(x0, x1), fmaxf(x2, x3));
#pragma unroll
      for (int o = 1; o < 16; o <<= 1) bm = fmaxf(bm, __shfl_xor(bm, o, 32));
      float mn = fmaxf(m_i[r], bm);
      float al = __expf(m_i[r] - mn);
      p[0][r] = __expf(x0 - mn);
      p[1][r] = __expf(x1 - mn);
      p[2][r] = __expf(x2 - mn);
      p[3][r] = __expf(x3 - mn);
      float rsum = (p[0][r] + p[1][r]) + (p[2][r] + p[3][r]);
#pragma unroll
      for (int o = 1; o < 16; o <<= 1) rsum += __shfl_xor(rsum, o, 32);
      l_i[r] = l_i[r] * al + rsum;
      m_i[r] = mn;
#pragma unroll
      for (int ht = 0; ht < 8; ++ht) accO[ht][r] *= al;
    }
    // ---- C-layout P -> LDS -> A-fragment layout (16x64 keys) ----
#pragma unroll
    for (int r = 0; r < 8; ++r) {
      int m = r + 8 * half;
#pragma unroll
      for (int t = 0; t < 4; ++t)
        lds_p[wave][m * 64 + t * 16 + ln] = (__bf16)p[t][r];
    }
    __syncthreads();
    v16bf pf[2];
#pragma unroll
    for (int kc = 0; kc < 2; ++kc) {
      const __bf16* pp = &lds_p[wave][ln * 64 + kc * 32 + half * 16];
      BF16x16 t;
      t.h[0] = *(const v8bf*)pp;
      t.h[1] = *(const v8bf*)(pp + 8);
      pf[kc] = t.v;
    }
    // ---- O += P * V over 8 h-subtiles x 2 key-chunks (16 WMMAs) ----
#pragma unroll
    for (int ht = 0; ht < 8; ++ht) {
#pragma unroll
      for (int kc = 0; kc < 2; ++kc) {
        const __bf16* vp = vtb + ((size_t)vh * H + ht * 16 + ln) * S + kbase +
                           kc * 32 + half * 16;
        v16bf bv = *(const v16bf*)vp;
        accO[ht] = wmma_bf16(pf[kc], bv, accO[ht]);
      }
    }
    __syncthreads();
  }
  // ---- normalize + store bf16 attn (S, U*H) ----
#pragma unroll
  for (int r = 0; r < 8; ++r) {
    float inv = 1.0f / l_i[r];
    int srow = q0 + r + 8 * half;
#pragma unroll
    for (int ht = 0; ht < 8; ++ht) {
      attnb[(size_t)srow * (U * H) + (size_t)u * H + ht * 16 + ln] =
          (__bf16)(accO[ht][r] * inv);
    }
  }
}

// ---------------- host launch ----------------
extern "C" void kernel_launch(void* const* d_in, const int* in_sizes, int n_in,
                              void* d_out, int out_size, void* d_ws, size_t ws_size,
                              hipStream_t stream) {
  (void)in_sizes; (void)n_in; (void)out_size; (void)ws_size;
  const float* hs  = (const float*)d_in[0];
  const int*   pos = (const int*)d_in[1];
  const float* Wq  = (const float*)d_in[2];
  const float* Wk  = (const float*)d_in[3];
  const float* Wv  = (const float*)d_in[4];
  const float* Wo  = (const float*)d_in[5];
  const float* qnw = (const float*)d_in[6];
  const float* knw = (const float*)d_in[7];
  float* out = (float*)d_out;

  char* w = (char*)d_ws;
  size_t off = 0;
  auto alloc = [&](size_t bytes) {
    void* p = w + off;
    off += (bytes + 255) & ~(size_t)255;
    return p;
  };
  __bf16* hsb  = (__bf16*)alloc((size_t)S * Mdim * 2);      // hidden bf16
  __bf16* wqp  = (__bf16*)alloc((size_t)Mdim * U * H * 2);  // packed Wq
  __bf16* wkp  = (__bf16*)alloc((size_t)Mdim * V * H * 2);
  __bf16* wvp  = (__bf16*)alloc((size_t)Mdim * V * H * 2);
  __bf16* wop  = (__bf16*)alloc((size_t)U * H * Mdim * 2);
  float*  qf   = (float*)alloc((size_t)S * U * H * 4);
  float*  kf   = (float*)alloc((size_t)S * V * H * 4);
  float*  vf   = (float*)alloc((size_t)S * V * H * 4);
  __bf16* qbuf = (__bf16*)alloc((size_t)S * U * H * 2);     // (S,U,H)
  __bf16* kbuf = (__bf16*)alloc((size_t)V * S * H * 2);     // (V,S,H)
  __bf16* vtb  = (__bf16*)alloc((size_t)V * H * S * 2);     // (V,H,S)
  __bf16* attb = (__bf16*)alloc((size_t)S * U * H * 2);     // (S,U*H)

  // 1) hidden fp32 -> bf16
  cvt_bf16_kernel<<<(S * Mdim) / 256, 256, 0, stream>>>(hs, hsb);
  // 2) pack weights into B-fragment layout
  pack_w_kernel<<<(Mdim / 32) * (U * H / 16) * 32 / 256, 256, 0, stream>>>(Wq, wqp, Mdim, U * H);
  pack_w_kernel<<<(Mdim / 32) * (V * H / 16) * 32 / 256, 256, 0, stream>>>(Wk, wkp, Mdim, V * H);
  pack_w_kernel<<<(Mdim / 32) * (V * H / 16) * 32 / 256, 256, 0, stream>>>(Wv, wvp, Mdim, V * H);
  pack_w_kernel<<<(U * H / 32) * (Mdim / 16) * 32 / 256, 256, 0, stream>>>(Wo, wop, U * H, Mdim);
  // 3) QKV projections (WMMA, async-LDS A staging)
  gemm_bf16_kernel<<<dim3(S / 128, (U * H) / 128), 128, 0, stream>>>(hsb, wqp, qf, Mdim, U * H);
  gemm_bf16_kernel<<<dim3(S / 128, (V * H) / 128), 128, 0, stream>>>(hsb, wkp, kf, Mdim, V * H);
  gemm_bf16_kernel<<<dim3(S / 128, (V * H) / 128), 128, 0, stream>>>(hsb, wvp, vf, Mdim, V * H);
  // 4) RMS-norm + mRoPE + layout shuffles
  qk_prep_kernel<<<(S * U) / 8, 256, 0, stream>>>(qf, qbuf, qnw, pos, U, 0);
  qk_prep_kernel<<<(S * V) / 8, 256, 0, stream>>>(kf, kbuf, knw, pos, V, 1);
  v_prep_kernel<<<(S * V * H) / 256, 256, 0, stream>>>(vf, vtb);
  // 5) flash attention
  flash_attn_kernel<<<dim3(S / 64, U), 128, 0, stream>>>(qbuf, kbuf, vtb, attb);
  // 6) output projection (WMMA) -> fp32 out
  gemm_bf16_kernel<<<dim3(S / 128, Mdim / 128), 128, 0, stream>>>(attb, wop, out, U * H, Mdim);
}
